// ModelGRU1_7146825581025
// MI455X (gfx1250) — matrix operand order repaired and checked by reading, not compiled, not run on hardware
//
#include <hip/hip_runtime.h>
#include <math.h>

typedef __attribute__((ext_vector_type(16))) _Float16 v16h;
typedef __attribute__((ext_vector_type(8)))  float    v8f;

#define B_  512
#define T_  512
#define E_  100
#define H_  40
#define NG  120              // 3H
#define M_  (B_*T_)          // 262144 GEMM rows
#define KP0 128              // padded K, layer-0 input (E=100 -> 128)
#define KP1 96               // padded K, layer-1 input (2H=80 -> 96)

union Frag16 { unsigned int u[8]; v16h h; };

// ---------------------------------------------------------------------------
// Embedding gather: X[b*T+t, 0:128] = emb[text[b,t], 0:100] ++ zeros  (f16)
// ---------------------------------------------------------------------------
__global__ void __launch_bounds__(256) embed_k(const int* __restrict__ text,
                                               const float* __restrict__ emb,
                                               _Float16* __restrict__ X)
{
    int i = blockIdx.x * 256 + threadIdx.x;       // grid = M_*KP0/256
    int r = i >> 7;                                // row
    int e = i & 127;                               // padded col
    int tok = text[r];
    float v = (e < E_) ? emb[(size_t)tok * E_ + e] : 0.f;
    X[i] = (_Float16)v;
}

// Zero the pad columns (80..95) of x1 rows (stride KP1)
__global__ void __launch_bounds__(256) zeropad_x1(_Float16* __restrict__ X)
{
    int i = blockIdx.x * 256 + threadIdx.x;       // grid = M_*16/256
    int row = i >> 4, c = 80 + (i & 15);
    X[(size_t)row * KP1 + c] = (_Float16)0.0f;
}

// ---------------------------------------------------------------------------
// Pack weight W[120,K] (f32) into WMMA B-fragment order (zero-padded):
//   dst[((kt*8+nt)*8+v)*32 + lane] = pair {W[n][k0], W[n][k0+1]} as 2xf16
//   n = nt*16 + lane%16 ; k0 = kt*32 + (v/4)*16 + (lane/16)*8 + 2*(v%4)
// ---------------------------------------------------------------------------
__global__ void __launch_bounds__(256) pack_w(const float* __restrict__ W,
                                              int K, int kiters,
                                              unsigned int* __restrict__ dst)
{
    int d = blockIdx.x * 256 + threadIdx.x;
    int total = kiters * 2048;
    if (d >= total) return;
    int lane = d & 31;
    int v    = (d >> 5) & 7;
    int nt   = (d >> 8) & 7;
    int kt   = d >> 11;
    int n    = nt * 16 + (lane & 15);
    int hi   = lane >> 4;
    int k0   = kt * 32 + (v >> 2) * 16 + hi * 8 + 2 * (v & 3);
    float lo = 0.f, hv = 0.f;
    if (n < NG && k0     < K) lo = W[(size_t)n * K + k0];
    if (n < NG && k0 + 1 < K) hv = W[(size_t)n * K + k0 + 1];
    union { _Float16 h[2]; unsigned int u; } pk;
    pk.h[0] = (_Float16)lo; pk.h[1] = (_Float16)hv;
    dst[d] = pk.u;
}

// ---------------------------------------------------------------------------
// Input-gate GEMM: C[M,120] = A[M,Kp](f16, zero-padded) * W^T + bias
// 8 waves/block, one 16-row M-tile per wave, all 8 N-tiles.
// A-fragment = two aligned 16-byte chunks per lane per k-iter (b128 loads).
// ---------------------------------------------------------------------------
__global__ void __launch_bounds__(256) gemm_gates(const _Float16* __restrict__ A,
                                                  int Kp, int kiters,
                                                  const unsigned int* __restrict__ pW,
                                                  const float* __restrict__ bias,
                                                  float* __restrict__ C)
{
    __shared__ unsigned int sW[4 * 2048];         // up to 32 KB of packed W
    int tid = threadIdx.x;
    int nW = kiters * 2048;
    for (int i = tid; i < nW; i += 256) sW[i] = pW[i];
    __syncthreads();

    int w = tid >> 5, lane = tid & 31, hi = lane >> 4, l = lane & 15;
    int mtile = blockIdx.x * 8 + w;
    int row = mtile * 16 + l;
    const _Float16* arow = A + (size_t)row * Kp;

    v8f acc[8] = {};
    for (int kt = 0; kt < kiters; ++kt) {
        Frag16 a;
        *reinterpret_cast<uint4*>(&a.u[0]) =
            *reinterpret_cast<const uint4*>(arow + kt * 32 + hi * 8);
        *reinterpret_cast<uint4*>(&a.u[4]) =
            *reinterpret_cast<const uint4*>(arow + kt * 32 + 16 + hi * 8);
#pragma unroll
        for (int nt = 0; nt < 8; ++nt) {
            Frag16 b;
#pragma unroll
            for (int v = 0; v < 8; ++v)
                b.u[v] = sW[(((kt * 8 + nt) * 8 + v) << 5) + lane];
            acc[nt] = __builtin_amdgcn_wmma_f32_16x16x32_f16(
                false, a.h, false, b.h, (short)0, acc[nt], false, false);
        }
    }
#pragma unroll
    for (int nt = 0; nt < 8; ++nt) {
        int col = nt * 16 + l;
        if (col < NG) {
#pragma unroll
            for (int v = 0; v < 8; ++v) {
                int orow = mtile * 16 + v + 8 * hi;
                C[(size_t)orow * NG + col] = acc[nt][v] + bias[col];
            }
        }
    }
}

// ---------------------------------------------------------------------------
// GRU recurrence. One block = 64-batch slab, one direction (blockIdx.y).
// Per step: gh = h(f16,LDS) @ Whh^T via 32 WMMA tiles, then gate math.
// gx is software-pipelined into registers (double buffer) + prefetched ahead.
// LAYER==0: writes x1[b,t, dir*40+j] (f16, stride 96). LAYER==1: fused pooling.
// ---------------------------------------------------------------------------
template <int LAYER>
__global__ void __launch_bounds__(256) gru_rec(
    const float* __restrict__ gxF, const float* __restrict__ gxB,
    const unsigned int* __restrict__ pWF, const unsigned int* __restrict__ pWB,
    const float* __restrict__ bhhF, const float* __restrict__ bhhB,
    const int* __restrict__ lens,
    _Float16* __restrict__ x1,
    float* __restrict__ hlastF, float* __restrict__ hlastB,
    float* __restrict__ avgp, float* __restrict__ maxp)
{
    __shared__ _Float16 h16[64][66];   // f16 state, cols 40..63 stay zero (K pad)
    __shared__ float    ghs[64][132];  // gh tile output
    __shared__ float    sbhh[120];
    __shared__ int      slens[64];

    int tid = threadIdx.x;
    int dir = blockIdx.y;
    int b0  = blockIdx.x * 64;
    const float* gx         = dir ? gxB : gxF;
    const unsigned int* pW  = dir ? pWB : pWF;
    const float* bhh        = dir ? bhhB : bhhF;
    float* hlast            = dir ? hlastB : hlastF;

    for (int i = tid; i < 64 * 66; i += 256) (&h16[0][0])[i] = (_Float16)0.0f;
    if (tid < 120) sbhh[tid] = bhh[tid];
    if (tid < 64)  slens[tid] = lens[b0 + tid];

    int w = tid >> 5, lane = tid & 31, hi = lane >> 4, l = lane & 15;

    // per-thread (b,j) pair ownership: pairs tid*10 .. tid*10+9
    int bl_[10], j_[10];
#pragma unroll
    for (int p = 0; p < 10; ++p) {
        int pi = tid * 10 + p;
        bl_[p] = pi / 40;
        j_[p]  = pi - bl_[p] * 40;
    }

    float hreg[10], psum[10], pmax[10];
#pragma unroll
    for (int p = 0; p < 10; ++p) { hreg[p] = 0.f; psum[p] = 0.f; pmax[p] = -__builtin_inff(); }

    float gxA[30], gxB2[30];
    auto ldgx = [&](int t, float (&buf)[30]) {
#pragma unroll
        for (int p = 0; p < 10; ++p) {
            size_t base = ((size_t)(b0 + bl_[p]) * T_ + t) * NG + j_[p];
            buf[p * 3 + 0] = gx[base];
            buf[p * 3 + 1] = gx[base + 40];
            buf[p * 3 + 2] = gx[base + 80];
        }
    };

    auto stepf = [&](int s, float (&use)[30], float (&nxt)[30]) {
        int t = dir ? (T_ - 1 - s) : s;
        __syncthreads();                       // h16 ready
        // issue next step's gx loads now: they overlap WMMA + barrier + gates
        if (s + 1 < T_) ldgx(dir ? (T_ - 2 - s) : (s + 1), nxt);
        // prefetch gx 8 steps ahead into cache (256 threads cover the slab)
        {
            int sp = s + 8;
            if (sp < T_) {
                int tp = dir ? (T_ - 1 - sp) : sp;
                const float* pfp = gx + ((size_t)(b0 + (tid >> 2)) * T_ + tp) * NG
                                      + (tid & 3) * 32;
                __builtin_prefetch(pfp, 0, 3);
            }
        }
        // ---- WMMA phase: 4 tiles per wave (4 M-tiles x 8 N-tiles total)
#pragma unroll
        for (int i = 0; i < 4; ++i) {
            int tile = w * 4 + i, mt = tile >> 3, nt = tile & 7;
            v8f acc = {};
#pragma unroll
            for (int kt = 0; kt < 2; ++kt) {
                Frag16 a, b;
#pragma unroll
                for (int v = 0; v < 8; ++v) {
                    int kb = kt * 32 + (v >> 2) * 16 + hi * 8 + ((v & 3) << 1);
                    a.u[v] = *reinterpret_cast<const unsigned int*>(&h16[mt * 16 + l][kb]);
                    b.u[v] = pW[(((kt * 8 + nt) * 8 + v) << 5) + lane];
                }
                acc = __builtin_amdgcn_wmma_f32_16x16x32_f16(
                    false, a.h, false, b.h, (short)0, acc, false, false);
            }
#pragma unroll
            for (int v = 0; v < 8; ++v)
                ghs[mt * 16 + v + 8 * hi][nt * 16 + l] = acc[v];
        }
        __syncthreads();                       // ghs ready
        // ---- elementwise gates: 10 (b,j) pairs per thread
#pragma unroll
        for (int p = 0; p < 10; ++p) {
            int bl = bl_[p], j = j_[p];
            int bg = b0 + bl;
            bool valid = t < slens[bl];
            float r = 1.f / (1.f + __expf(-(use[p * 3 + 0] + ghs[bl][j]      + sbhh[j])));
            float z = 1.f / (1.f + __expf(-(use[p * 3 + 1] + ghs[bl][40 + j] + sbhh[40 + j])));
            float n = tanhf(use[p * 3 + 2] + r * (ghs[bl][80 + j] + sbhh[80 + j]));
            float hp = hreg[p];
            float hc = (1.f - z) * n + z * hp;
            float hnew = valid ? hc : hp;
            float outv = valid ? hc : 0.f;
            hreg[p] = hnew;
            h16[bl][j] = (_Float16)hnew;
            if (LAYER == 0) {
                x1[((size_t)bg * T_ + t) * KP1 + dir * H_ + j] = (_Float16)outv;
            } else {
                psum[p] += outv;
                if (valid && hc > pmax[p]) pmax[p] = hc;
            }
        }
    };

    ldgx(dir ? (T_ - 1) : 0, gxA);
    for (int s2 = 0; s2 < T_; s2 += 2) {
        stepf(s2,     gxA,  gxB2);
        stepf(s2 + 1, gxB2, gxA);
    }

    // ---- epilogue: h_last + pools
#pragma unroll
    for (int p = 0; p < 10; ++p) {
        int bg = b0 + bl_[p], j = j_[p];
        hlast[(size_t)bg * H_ + j] = hreg[p];
        if (LAYER == 1) {
            float len = (float)slens[bl_[p]];
            avgp[(size_t)bg * (2 * H_) + dir * H_ + j] = psum[p] / len;
            maxp[(size_t)bg * (2 * H_) + dir * H_ + j] = pmax[p];
        }
    }
}

// ---------------------------------------------------------------------------
// Head: pool_cat=[hb1,hf1,hb0,hf0,avg,max] -> fc1 -> leaky_relu -> fc2
// ---------------------------------------------------------------------------
__global__ void __launch_bounds__(128) fc_head(const float* __restrict__ hb1,
                                               const float* __restrict__ hf1,
                                               const float* __restrict__ hb0,
                                               const float* __restrict__ hf0,
                                               const float* __restrict__ avgp,
                                               const float* __restrict__ maxp,
                                               const float* __restrict__ fc1W,
                                               const float* __restrict__ fc1b,
                                               const float* __restrict__ fc2W,
                                               const float* __restrict__ fc2b,
                                               float* __restrict__ out)
{
    int b = blockIdx.x, j = threadIdx.x;
    __shared__ float pc[320];
    __shared__ float red[128];
    for (int i = j; i < 320; i += 128) {
        float v;
        if      (i < 40)  v = hb1[b * 40 + i];
        else if (i < 80)  v = hf1[b * 40 + i - 40];
        else if (i < 120) v = hb0[b * 40 + i - 80];
        else if (i < 160) v = hf0[b * 40 + i - 120];
        else if (i < 240) v = avgp[b * 80 + i - 160];
        else              v = maxp[b * 80 + i - 240];
        pc[i] = v;
    }
    __syncthreads();
    float acc = fc1b[j];
    for (int i = 0; i < 320; ++i) acc += fc1W[j * 320 + i] * pc[i];
    float h = (acc >= 0.f) ? acc : 0.01f * acc;
    red[j] = h * fc2W[j];
    __syncthreads();
    for (int s = 64; s > 0; s >>= 1) {
        if (j < s) red[j] += red[j + s];
        __syncthreads();
    }
    if (j == 0) out[b] = red[0] + fc2b[0];
}

// ---------------------------------------------------------------------------
extern "C" void kernel_launch(void* const* d_in, const int* in_sizes, int n_in,
                              void* d_out, int out_size, void* d_ws, size_t ws_size,
                              hipStream_t stream)
{
    (void)in_sizes; (void)n_in; (void)out_size; (void)ws_size;
    const int*   text  = (const int*)d_in[0];
    const int*   tlen  = (const int*)d_in[1];
    const float* emb   = (const float*)d_in[2];
    const float* Wih0f = (const float*)d_in[3];
    const float* Whh0f = (const float*)d_in[4];
    const float* bih0f = (const float*)d_in[5];
    const float* bhh0f = (const float*)d_in[6];
    const float* Wih0b = (const float*)d_in[7];
    const float* Whh0b = (const float*)d_in[8];
    const float* bih0b = (const float*)d_in[9];
    const float* bhh0b = (const float*)d_in[10];
    const float* Wih1f = (const float*)d_in[11];
    const float* Whh1f = (const float*)d_in[12];
    const float* bih1f = (const float*)d_in[13];
    const float* bhh1f = (const float*)d_in[14];
    const float* Wih1b = (const float*)d_in[15];
    const float* Whh1b = (const float*)d_in[16];
    const float* bih1b = (const float*)d_in[17];
    const float* bhh1b = (const float*)d_in[18];
    const float* fc1W  = (const float*)d_in[19];
    const float* fc1b  = (const float*)d_in[20];
    const float* fc2W  = (const float*)d_in[21];
    const float* fc2b  = (const float*)d_in[22];

    char* ws = (char*)d_ws;
    _Float16* X = (_Float16*)(ws + 0);              //  67,108,864 B (X pad128 / x1 pad96)
    float* G0 = (float*)(ws + 67108864);            // 125,829,120 B (gx fwd)
    float* G1 = (float*)(ws + 192937984);           // 125,829,120 B (gx bwd)
    char* sm = ws + 318767104;
    unsigned int* pWih0f = (unsigned int*)(sm + 0);
    unsigned int* pWih0b = (unsigned int*)(sm + 32768);
    unsigned int* pWih1f = (unsigned int*)(sm + 65536);
    unsigned int* pWih1b = (unsigned int*)(sm + 90112);
    unsigned int* pWhh0f = (unsigned int*)(sm + 114688);
    unsigned int* pWhh0b = (unsigned int*)(sm + 131072);
    unsigned int* pWhh1f = (unsigned int*)(sm + 147456);
    unsigned int* pWhh1b = (unsigned int*)(sm + 163840);
    float* hf0  = (float*)(sm + 180224);
    float* hb0  = (float*)(sm + 262144);
    float* hf1  = (float*)(sm + 344064);
    float* hb1  = (float*)(sm + 425984);
    float* avgp = (float*)(sm + 507904);
    float* maxp = (float*)(sm + 671744);

    // pack weights to WMMA B-fragment order (f16, zero-padded K)
    pack_w<<<32, 256, 0, stream>>>(Wih0f, 100, 4, pWih0f);
    pack_w<<<32, 256, 0, stream>>>(Wih0b, 100, 4, pWih0b);
    pack_w<<<24, 256, 0, stream>>>(Wih1f,  80, 3, pWih1f);
    pack_w<<<24, 256, 0, stream>>>(Wih1b,  80, 3, pWih1b);
    pack_w<<<16, 256, 0, stream>>>(Whh0f,  40, 2, pWhh0f);
    pack_w<<<16, 256, 0, stream>>>(Whh0b,  40, 2, pWhh0b);
    pack_w<<<16, 256, 0, stream>>>(Whh1f,  40, 2, pWhh1f);
    pack_w<<<16, 256, 0, stream>>>(Whh1b,  40, 2, pWhh1b);

    // embedding gather (f16, row stride 128 with zero pad)
    embed_k<<<M_ * KP0 / 256, 256, 0, stream>>>(text, emb, X);

    // layer 0: input gates (both directions), then recurrence (writes x1 over X)
    gemm_gates<<<2048, 256, 0, stream>>>(X, KP0, 4, pWih0f, bih0f, G0);
    gemm_gates<<<2048, 256, 0, stream>>>(X, KP0, 4, pWih0b, bih0b, G1);
    zeropad_x1<<<M_ * 16 / 256, 256, 0, stream>>>(X);
    gru_rec<0><<<dim3(8, 2), 256, 0, stream>>>(G0, G1, pWhh0f, pWhh0b, bhh0f, bhh0b,
                                               tlen, X, hf0, hb0, nullptr, nullptr);

    // layer 1: input gates on x1 (stride 96), recurrence with fused pooling
    gemm_gates<<<2048, 256, 0, stream>>>(X, KP1, 3, pWih1f, bih1f, G0);
    gemm_gates<<<2048, 256, 0, stream>>>(X, KP1, 3, pWih1b, bih1b, G1);
    gru_rec<1><<<dim3(8, 2), 256, 0, stream>>>(G0, G1, pWhh1f, pWhh1b, bhh1f, bhh1b,
                                               tlen, nullptr, hf1, hb1, avgp, maxp);

    // head
    fc_head<<<512, 128, 0, stream>>>(hb1, hf1, hb0, hf0, avgp, maxp,
                                     fc1W, fc1b, fc2W, fc2b, (float*)d_out);
}
